// ALIGNNConvPyG_919123001698
// MI455X (gfx1250) — matrix operand order repaired and checked by reading, not compiled, not run on hardware
//
#include <hip/hip_runtime.h>
#include <hip/hip_bf16.h>
#include <stdint.h>

#define F_DIM 128
#define NN    50000
#define NE    400000
#define NLE   800000

typedef __bf16 bf16_t;
typedef __attribute__((ext_vector_type(16))) __bf16 v16bf;
typedef __attribute__((ext_vector_type(8)))  float  v8f;

// native bf16 convert -> v_cvt_pk_bf16_f32 (verified in asm)
__device__ __forceinline__ bf16_t f2bf(float x) { return (bf16_t)x; }

__device__ __forceinline__ float sigmoidf_(float x) {
  return 1.0f / (1.0f + __expf(-x));
}

__device__ __forceinline__ v8f wmma_bf16(v16bf a, v16bf b, v8f c) {
  return __builtin_amdgcn_wmma_f32_16x16x32_bf16(false, a, false, b, (short)0, c, false, false);
}

// opaque zero: breaks loop-invariance so LDS B-fragment loads are NOT hoisted
// out of the tile loop (round-3 asm showed them spilled to scratch otherwise)
__device__ __forceinline__ int opaque_zero() {
  int v = 0;
  asm volatile("" : "+v"(v));
  return v;
}

// A-fragment (16x32 bf16) per ISA layout: lane holds row (lane&15);
// elements 0..7 -> K = k0+kb+0..7 ; elements 8..15 -> K = k0+kb+16..23, kb = (lane>=16)*8
__device__ __forceinline__ v16bf load_a_frag(const float* __restrict__ row, int k0, int kb) {
  const float4* p = (const float4*)(row + k0 + kb);
  const float4* q = (const float4*)(row + k0 + kb + 16);
  float4 f0 = p[0], f1 = p[1], f2 = q[0], f3 = q[1];
  v16bf a;
  a[0] = f2bf(f0.x); a[1] = f2bf(f0.y); a[2]  = f2bf(f0.z); a[3]  = f2bf(f0.w);
  a[4] = f2bf(f1.x); a[5] = f2bf(f1.y); a[6]  = f2bf(f1.z); a[7]  = f2bf(f1.w);
  a[8] = f2bf(f2.x); a[9] = f2bf(f2.y); a[10] = f2bf(f2.z); a[11] = f2bf(f2.w);
  a[12] = f2bf(f3.x); a[13] = f2bf(f3.y); a[14] = f2bf(f3.z); a[15] = f2bf(f3.w);
  return a;
}

__device__ __forceinline__ v16bf load_b_frag(const bf16_t* p) {
  typedef __attribute__((ext_vector_type(8))) __bf16 v8bf;
  v8bf lo = *(const v8bf*)p;
  v8bf hi = *(const v8bf*)(p + 16);
  return __builtin_shufflevector(lo, hi, 0, 1, 2, 3, 4, 5, 6, 7, 8, 9, 10, 11, 12, 13, 14, 15);
}

// ---------------------------------------------------------------- utility
__global__ void zero_kernel(float* __restrict__ p, long long n) {
  long long i = (long long)blockIdx.x * blockDim.x + threadIdx.x;
  long long stride = (long long)gridDim.x * blockDim.x;
  for (; i < n; i += stride) p[i] = 0.0f;
}

// W[5][128][128] row-major (out_f, k)  ->  WT[5][128][128] row-major (out_f, k) in bf16
__global__ void prep_weights(const float* __restrict__ W1, const float* __restrict__ W2,
                             bf16_t* __restrict__ WT1, bf16_t* __restrict__ WT2) {
  int i = blockIdx.x * blockDim.x + threadIdx.x;
  const int total = 5 * F_DIM * F_DIM;
  if (i < total) {
    int w = i >> 14, r = i & 16383, k = r >> 7, f = r & 127;
    int src = i; (void)w; (void)k; (void)f;
    WT1[i] = f2bf(W1[src]);
    WT2[i] = f2bf(W2[src]);
  }
}

// ---------------------------------------------------------------- node GEMM: out = X @ W3^T + b3
__global__ void __launch_bounds__(256, 4) node_gemm_kernel(
    const float* __restrict__ X, const bf16_t* __restrict__ WT3,
    const float* __restrict__ bias3, float* __restrict__ out,
    int Nrows, int ntiles) {
  extern __shared__ char smem0[];
  bf16_t* sW = (bf16_t*)smem0;  // 128x128 bf16 = 32 KB
  for (int t = threadIdx.x; t < 2048; t += blockDim.x)
    ((uint4*)sW)[t] = ((const uint4*)WT3)[t];
  __syncthreads();

  const int lane = threadIdx.x & 31, wave = threadIdx.x >> 5;
  const int col = lane & 15;
  const int kb = (lane >> 4) * 8;
  const int hi8 = (lane >> 4) * 8;
  float bcol[8];
#pragma unroll
  for (int nt = 0; nt < 8; nt++) bcol[nt] = bias3[nt * 16 + col];

  const int wavesTotal = gridDim.x * (blockDim.x >> 5);
  for (int tile = blockIdx.x * (blockDim.x >> 5) + wave; tile < ntiles; tile += wavesTotal) {
    const int soff = opaque_zero();  // anti-LICM for LDS B loads
    int r0 = tile * 16;
    int rr = r0 + col; if (rr >= Nrows) rr = Nrows - 1;
    const float* row = X + (size_t)rr * F_DIM;
    v8f acc[8];
#pragma unroll
    for (int nt = 0; nt < 8; nt++) {
#pragma unroll
      for (int v = 0; v < 8; v++) acc[nt][v] = bcol[nt];
    }
#pragma unroll
    for (int ks = 0; ks < 4; ks++) {
      int k0 = ks * 32;
      v16bf a = load_a_frag(row, k0, kb);
      int koff = col * F_DIM + k0 + kb + soff;
#pragma unroll
      for (int nt = 0; nt < 8; nt++) {
        v16bf b = load_b_frag(sW + koff + nt * 16 * F_DIM);
        acc[nt] = wmma_bf16(a, b, acc[nt]);
      }
    }
#pragma unroll
    for (int v = 0; v < 8; v++) {
      int M = v + hi8, rw = r0 + M;
      if (rw < Nrows) {
        float* orow = out + (size_t)rw * F_DIM;
#pragma unroll
        for (int nt = 0; nt < 8; nt++) orow[nt * 16 + col] = acc[nt][v];
      }
    }
  }
}

// ------------------------------------------------- edge kernel, two K-passes to stay <256 VGPRs:
//   pass 1: msg = xi@W0^T + xj@W1^T + e@W2^T + b012 ; stores, sigmoid segment-sum, BN stats
//   pass 2: upd = xj@W4^T + b4 ; stores
__global__ void __launch_bounds__(256, 4) egconv_edge_kernel(
    const int* __restrict__ idx_i, const int* __restrict__ idx_j,
    const float* __restrict__ X, const float* __restrict__ Efeat,
    const bf16_t* __restrict__ WT, const float* __restrict__ bias,
    float* __restrict__ msg, float* __restrict__ upd,
    float* __restrict__ sig_sum, float* __restrict__ stats,
    int E, int ntiles) {
  extern __shared__ char smem1[];
  bf16_t* sW = (bf16_t*)smem1;  // 4 matrices {W0,W1,W2,W4}, 128 KB
  for (int t = threadIdx.x; t < 8192; t += blockDim.x) {
    int wi = t >> 11;
    int w = (wi == 3) ? 4 : wi;
    ((uint4*)sW)[t] = ((const uint4*)(WT + (size_t)w * 16384))[t & 2047];
  }
  __syncthreads();

  const int lane = threadIdx.x & 31, wave = threadIdx.x >> 5;
  const int col = lane & 15;
  const int kb = (lane >> 4) * 8;
  const int hi8 = (lane >> 4) * 8;

  float bmsg[8], bupd[8], fsum[8], fsq[8];
#pragma unroll
  for (int nt = 0; nt < 8; nt++) {
    int f = nt * 16 + col;
    bmsg[nt] = bias[f] + bias[F_DIM + f] + bias[2 * F_DIM + f];
    bupd[nt] = bias[4 * F_DIM + f];
    fsum[nt] = 0.0f; fsq[nt] = 0.0f;
  }

  const int wavesTotal = gridDim.x * (blockDim.x >> 5);
  for (int tile = blockIdx.x * (blockDim.x >> 5) + wave; tile < ntiles; tile += wavesTotal) {
    const int soff = opaque_zero();  // anti-LICM for LDS B loads
    const bf16_t* sW0 = sW + soff;
    const bf16_t* sW1 = sW + 16384 + soff;
    const bf16_t* sW2 = sW + 32768 + soff;
    const bf16_t* sW4 = sW + 49152 + soff;

    int e0 = tile * 16;
    int er = e0 + col; if (er >= E) er = E - 1;
    int gi = idx_i[er], gj = idx_j[er];
    const float* xi = X + (size_t)gi * F_DIM;
    const float* xj = X + (size_t)gj * F_DIM;
    const float* ef = Efeat + (size_t)er * F_DIM;

    // prefetch next tile's gather rows while this tile computes (global_prefetch_b8)
    {
      int tn = tile + wavesTotal;
      if (tn < ntiles) {
        int en = tn * 16 + col; if (en >= E) en = E - 1;
        int gin = idx_i[en], gjn = idx_j[en];
        const char* pxi = (const char*)(X + (size_t)gin * F_DIM);
        const char* pxj = (const char*)(X + (size_t)gjn * F_DIM);
        const char* pef = (const char*)(Efeat + (size_t)en * F_DIM);
        __builtin_prefetch(pxi, 0, 1); __builtin_prefetch(pxi + 256, 0, 1);
        __builtin_prefetch(pxj, 0, 1); __builtin_prefetch(pxj + 256, 0, 1);
        __builtin_prefetch(pef, 0, 1); __builtin_prefetch(pef + 256, 0, 1);
      }
    }

    // ================= pass 1 : edge_msg =================
    {
      v8f accM[8];
#pragma unroll
      for (int nt = 0; nt < 8; nt++) {
#pragma unroll
        for (int v = 0; v < 8; v++) accM[nt][v] = bmsg[nt];
      }
#pragma unroll
      for (int ks = 0; ks < 4; ks++) {
        int k0 = ks * 32;
        v16bf a_i = load_a_frag(xi, k0, kb);
        v16bf a_j = load_a_frag(xj, k0, kb);
        v16bf a_e = load_a_frag(ef, k0, kb);
        int koff = col * F_DIM + k0 + kb;
#pragma unroll
        for (int nt = 0; nt < 8; nt++) {
          int off = koff + nt * 16 * F_DIM;
          v16bf b0 = load_b_frag(sW0 + off);
          v16bf b1 = load_b_frag(sW1 + off);
          v16bf b2 = load_b_frag(sW2 + off);
          accM[nt] = wmma_bf16(a_i, b0, accM[nt]);
          accM[nt] = wmma_bf16(a_j, b1, accM[nt]);
          accM[nt] = wmma_bf16(a_e, b2, accM[nt]);
        }
      }
      // epilogue: D element = (M = v + hi8, N = nt*16 + col)
#pragma unroll
      for (int v = 0; v < 8; v++) {
        int M = v + hi8;
        int e = e0 + M;
        int giv = __shfl(gi, M, 32);  // lane M holds idx_i of row M
        if (e < E) {
          float* mrow = msg + (size_t)e * F_DIM;
          float* srow = sig_sum + (size_t)giv * F_DIM;
#pragma unroll
          for (int nt = 0; nt < 8; nt++) {
            int f = nt * 16 + col;
            float mv = accM[nt][v];
            mrow[f] = mv;
            atomicAdd(srow + f, sigmoidf_(mv));
            fsum[nt] += mv;
            fsq[nt] += mv * mv;
          }
        }
      }
    }

    // ================= pass 2 : upd =================
    {
      v8f accU[8];
#pragma unroll
      for (int nt = 0; nt < 8; nt++) {
#pragma unroll
        for (int v = 0; v < 8; v++) accU[nt][v] = bupd[nt];
      }
#pragma unroll
      for (int ks = 0; ks < 4; ks++) {
        int k0 = ks * 32;
        v16bf a_j = load_a_frag(xj, k0, kb);
        int koff = col * F_DIM + k0 + kb;
#pragma unroll
        for (int nt = 0; nt < 8; nt++) {
          v16bf b4 = load_b_frag(sW4 + koff + nt * 16 * F_DIM);
          accU[nt] = wmma_bf16(a_j, b4, accU[nt]);
        }
      }
#pragma unroll
      for (int v = 0; v < 8; v++) {
        int M = v + hi8;
        int e = e0 + M;
        if (e < E) {
          float* urow = upd + (size_t)e * F_DIM;
#pragma unroll
          for (int nt = 0; nt < 8; nt++) urow[nt * 16 + col] = accU[nt][v];
        }
      }
    }
  }
#pragma unroll
  for (int nt = 0; nt < 8; nt++) {
    int f = nt * 16 + col;
    atomicAdd(stats + f, fsum[nt]);
    atomicAdd(stats + F_DIM + f, fsq[nt]);
  }
}

// ----------------------------------------- aggregate: agg[i] += sigmoid(msg)*upd/(sigsum[i]+eps)
__global__ void aggregate_kernel(const int* __restrict__ idx_i,
                                 const float* __restrict__ msg, const float* __restrict__ upd,
                                 const float* __restrict__ sig_sum, float* __restrict__ agg, int E) {
  long long q = (long long)blockIdx.x * blockDim.x + threadIdx.x;
  long long total = (long long)E * 32;  // float4 granularity
  long long stride = (long long)gridDim.x * blockDim.x;
  for (; q < total; q += stride) {
    int e = (int)(q >> 5), c4 = (int)(q & 31);
    float4 m = ((const float4*)msg)[q];
    float4 u = ((const float4*)upd)[q];
    int i = idx_i[e];
    float4 s4 = *(const float4*)(sig_sum + (size_t)i * F_DIM + c4 * 4);
    float* a = agg + (size_t)i * F_DIM + c4 * 4;
    atomicAdd(a + 0, sigmoidf_(m.x) * u.x / (s4.x + 1e-9f));
    atomicAdd(a + 1, sigmoidf_(m.y) * u.y / (s4.y + 1e-9f));
    atomicAdd(a + 2, sigmoidf_(m.z) * u.z / (s4.z + 1e-9f));
    atomicAdd(a + 3, sigmoidf_(m.w) * u.w / (s4.w + 1e-9f));
  }
}

// ----------------------------------------- per-feature sum / sumsq over rows
__global__ void feat_stats_kernel(const float* __restrict__ h, float* __restrict__ stats, int rows) {
  int f = threadIdx.x;  // blockDim = 128
  float s = 0.0f, q = 0.0f;
  for (int r = blockIdx.x; r < rows; r += gridDim.x) {
    float v = h[(size_t)r * F_DIM + f];
    s += v; q += v * v;
  }
  atomicAdd(stats + f, s);
  atomicAdd(stats + F_DIM + f, q);
}

// ----------------------------------------- out = base + silu(BN(h))
__global__ void update_out_kernel(const float* __restrict__ base, const float* __restrict__ h,
                                  const float* __restrict__ stats, const float* __restrict__ gamma,
                                  const float* __restrict__ beta, float* __restrict__ out,
                                  long long n, float invCount) {
  long long i = (long long)blockIdx.x * blockDim.x + threadIdx.x;
  long long stride = (long long)gridDim.x * blockDim.x;
  for (; i < n; i += stride) {
    int f = (int)(i & 127);
    float mu = stats[f] * invCount;
    float var = stats[F_DIM + f] * invCount - mu * mu;
    float inv = rsqrtf(var + 1e-5f);
    float bnv = gamma[f] * (h[i] - mu) * inv + beta[f];
    out[i] = base[i] + bnv * sigmoidf_(bnv);
  }
}

extern "C" void kernel_launch(void* const* d_in, const int* in_sizes, int n_in,
                              void* d_out, int out_size, void* d_ws, size_t ws_size,
                              hipStream_t stream) {
  const int*   g_ei  = (const int*)d_in[0];
  const int*   lg_ei = (const int*)d_in[1];
  const float* x     = (const float*)d_in[2];
  const float* y     = (const float*)d_in[3];
  const float* z     = (const float*)d_in[4];
  const float* W1    = (const float*)d_in[5];
  const float* b1    = (const float*)d_in[6];
  const float* bng1  = (const float*)d_in[7];
  const float* bnb1  = (const float*)d_in[8];
  const float* W2    = (const float*)d_in[9];
  const float* b2    = (const float*)d_in[10];
  const float* bng2  = (const float*)d_in[11];
  const float* bnb2  = (const float*)d_in[12];

  float* out_x = (float*)d_out;
  float* out_y = out_x + (size_t)NN * F_DIM;
  float* out_z = out_y + (size_t)NE * F_DIM;

  char* base = (char*)d_ws;
  bf16_t* WT1   = (bf16_t*)(base);             // 160 KB
  bf16_t* WT2   = (bf16_t*)(base + 163840);    // 160 KB
  float*  STATS = (float*)(base + 327680);     // 1024 floats: [E1 sum/sq][N1][E2][N2]
  char*   SCR   = base + 331776;
  // layer-1 scratch; layer-2 scratch overlaps it (layer 1 fully done first)
  float* MSG1 = (float*)(SCR);
  float* UPD1 = (float*)(SCR + 204800000ll);
  float* SIG1 = (float*)(SCR + 409600000ll);
  float* AGG1 = (float*)(SCR + 435200000ll);
  float* MSG2 = (float*)(SCR);
  float* UPD2 = (float*)(SCR + 409600000ll);
  float* SIG2 = (float*)(SCR + 819200000ll);   // outside layer-1 footprint -> safe to pre-zero
  float* AGG2 = (float*)(SCR + 1024000000ll);

  zero_kernel<<<2048, 256, 0, stream>>>(SIG1, (long long)NN * F_DIM);
  zero_kernel<<<4096, 256, 0, stream>>>(SIG2, (long long)NE * F_DIM);
  zero_kernel<<<4, 256, 0, stream>>>(STATS, 1024);
  prep_weights<<<(5 * F_DIM * F_DIM + 255) / 256, 256, 0, stream>>>(W1, W2, WT1, WT2);

  // -------- layer 1 : graph g, node feats x, edge feats y --------
  {
    int ntilesN = (NN + 15) / 16;
    int nblkN = (ntilesN + 7) / 8; if (nblkN > 512) nblkN = 512;
    node_gemm_kernel<<<nblkN, 256, 32768, stream>>>(x, WT1 + 3 * 16384, b1 + 3 * F_DIM,
                                                    AGG1, NN, ntilesN);
    int ntilesE = (NE + 15) / 16;
    int nblkE = (ntilesE + 7) / 8; if (nblkE > 512) nblkE = 512;
    egconv_edge_kernel<<<nblkE, 256, 131072, stream>>>(g_ei, g_ei + NE, x, y, WT1, b1,
                                                       MSG1, UPD1, SIG1, STATS + 0, NE, ntilesE);
    aggregate_kernel<<<4096, 256, 0, stream>>>(g_ei, MSG1, UPD1, SIG1, AGG1, NE);
    feat_stats_kernel<<<1024, 128, 0, stream>>>(AGG1, STATS + 256, NN);
    update_out_kernel<<<8192, 256, 0, stream>>>(y, MSG1, STATS + 0, bng1, bnb1,
                                                out_y, (long long)NE * F_DIM, 1.0f / NE);  // m
    update_out_kernel<<<2048, 256, 0, stream>>>(x, AGG1, STATS + 256, bng1 + F_DIM, bnb1 + F_DIM,
                                                out_x, (long long)NN * F_DIM, 1.0f / NN);
  }
  // -------- layer 2 : line graph, node feats m (= out_y), edge feats z --------
  {
    int ntilesN = (NE + 15) / 16;
    int nblkN = (ntilesN + 7) / 8; if (nblkN > 512) nblkN = 512;
    node_gemm_kernel<<<nblkN, 256, 32768, stream>>>(out_y, WT2 + 3 * 16384, b2 + 3 * F_DIM,
                                                    AGG2, NE, ntilesN);
    int ntilesE = (NLE + 15) / 16;
    int nblkE = (ntilesE + 7) / 8; if (nblkE > 512) nblkE = 512;
    egconv_edge_kernel<<<nblkE, 256, 131072, stream>>>(lg_ei, lg_ei + NLE, out_y, z, WT2, b2,
                                                       MSG2, UPD2, SIG2, STATS + 512, NLE, ntilesE);
    aggregate_kernel<<<8192, 256, 0, stream>>>(lg_ei, MSG2, UPD2, SIG2, AGG2, NLE);
    feat_stats_kernel<<<1024, 128, 0, stream>>>(AGG2, STATS + 768, NE);
    update_out_kernel<<<8192, 256, 0, stream>>>(z, MSG2, STATS + 512, bng2, bnb2,
                                                out_z, (long long)NLE * F_DIM, 1.0f / NLE);
    update_out_kernel<<<8192, 256, 0, stream>>>(out_y, AGG2, STATS + 768, bng2 + F_DIM, bnb2 + F_DIM,
                                                out_y, (long long)NE * F_DIM, 1.0f / NE);
  }
  (void)in_sizes; (void)n_in; (void)out_size; (void)ws_size;
}
